// StyleMod_31490700214661
// MI455X (gfx1250) — matrix operand order repaired and checked by reading, not compiled, not run on hardware
//
#include <hip/hip_runtime.h>

// ---------------------------------------------------------------------------
// StyleGAN2 modulated 3x3 conv for MI455X (gfx1250, wave32, WMMA bf16 + TDM).
//   Stage 1: style = w @ style_w^T + style_b          (16x256, K=512)
//   Stage 2: wt = weight*SCALE*(style+1); demod; pack bf16 [b][o][tap][ch]
//   Stage 3: pack x -> zero-padded NHWC bf16  [b][66][66][256]
//   Stage 4: implicit-GEMM conv: TENSOR_LOAD_TO_LDS double-buffered staging
//            + V_WMMA_F32_16X16X32_BF16 inner loop.
// Workspace: 16KB style + 18.9MB weights + 34.1MB packed x  (~52.6MB total)
// ---------------------------------------------------------------------------

#define IN_CH   256
#define OUT_CH  256
#define LATENT  512
#define KTAPS   9
#define IMG     64
#define NPIX    (IMG * IMG)
#define BATCH   16
#define PAD     66                       // 64 + 2 halo (zero-padded)

#define TILE_ELEMS (3 * 66 * 32)         // rows x cols x channels per K-block
#define TILE_BYTES (TILE_ELEMS * 2)      // 12672 B bf16

typedef __attribute__((ext_vector_type(16))) __bf16 v16bf;
typedef __attribute__((ext_vector_type(8)))  __bf16 v8bf;
typedef __attribute__((ext_vector_type(8)))  float  v8f;

// ---------------------------------------------------------------------------
// Kernel 1: style factor  sp[b][i] = SCALE * (dot(w[b], style_w[i]) + b + 1)
// ---------------------------------------------------------------------------
__global__ __launch_bounds__(256)
void style_factor_kernel(const float* __restrict__ w,
                         const float* __restrict__ style_w,
                         const float* __restrict__ style_b,
                         float* __restrict__ sp) {
    const int b = blockIdx.x;      // 0..15
    const int i = threadIdx.x;     // 0..255
    const float scale = 1.0f / 16.0f;   // 1/sqrt(IN_CH)
    float acc = 0.0f;
    const float* wrow = w + b * LATENT;
    const float* srow = style_w + (size_t)i * LATENT;
    for (int l = 0; l < LATENT; ++l)
        acc = fmaf(wrow[l], srow[l], acc);
    sp[b * IN_CH + i] = scale * (acc + style_b[i] + 1.0f);
}

// ---------------------------------------------------------------------------
// Kernel 2: modulate + demod + pack to bf16.
// Layout: Wbf[((b*256 + o)*9 + t)*256 + i]   (in-channels contiguous)
// ---------------------------------------------------------------------------
__global__ __launch_bounds__(256)
void modulate_weight_kernel(const float* __restrict__ weight,
                            const float* __restrict__ sp,
                            __bf16* __restrict__ Wbf) {
    const int bo = blockIdx.x;          // 0..4095
    const int b  = bo >> 8;
    const int o  = bo & 255;
    const int i  = threadIdx.x;

    const float s = sp[b * IN_CH + i];
    const float* wp = weight + ((size_t)o * IN_CH + i) * KTAPS;

    float wt[KTAPS];
    float local = 0.0f;
    for (int t = 0; t < KTAPS; ++t) {
        float v = wp[t] * s;
        wt[t] = v;
        local = fmaf(v, v, local);
    }

    __shared__ float red[256];
    red[i] = local;
    __syncthreads();
    for (int st = 128; st > 0; st >>= 1) {
        if (i < st) red[i] += red[i + st];
        __syncthreads();
    }
    const float demod = rsqrtf(red[0] + 1e-8f);

    __bf16* dst = Wbf + ((size_t)(b * OUT_CH + o) * KTAPS) * IN_CH + i;
    for (int t = 0; t < KTAPS; ++t)
        dst[t * IN_CH] = (__bf16)(wt[t] * demod);
}

// ---------------------------------------------------------------------------
// Kernel 3: pack x into zero-padded NHWC bf16:  xpad[b][p][col][c]
//   p = image row + 1, col = image col + 1; borders (p/col == 0 or 65) = 0.
// ---------------------------------------------------------------------------
__global__ __launch_bounds__(256)
void pack_x_kernel(const float* __restrict__ x, __bf16* __restrict__ xpad) {
    const int p = blockIdx.x;      // 0..65 padded row
    const int b = blockIdx.y;
    const int c = threadIdx.x;     // 0..255
    __bf16* orow = xpad + ((size_t)(b * PAD + p) * PAD) * IN_CH + c;
    const int y = p - 1;
    if (y < 0 || y >= IMG) {
        for (int col = 0; col < PAD; ++col) orow[(size_t)col * IN_CH] = (__bf16)0.0f;
        return;
    }
    const float* xrow = x + ((size_t)(b * IN_CH + c) * IMG + y) * IMG;
    orow[0] = (__bf16)0.0f;
    for (int xx = 0; xx < IMG; ++xx)
        orow[(size_t)(xx + 1) * IN_CH] = (__bf16)xrow[xx];
    orow[(size_t)(PAD - 1) * IN_CH] = (__bf16)0.0f;
}

// ---------------------------------------------------------------------------
// TDM: DMA one [3 rows][66 cols][32 ch] bf16 tile from packed x into LDS.
// D# per CDNA5 ISA:
//   tile  X=32 ch (contig), Y=66 cols (stride 256), Z=3 rows (stride 66*256)
//   data_size=2B, count=1, type=2.
// lds_addr: low 32 bits of the buffer's flat address (= wave-relative LDS
// byte offset per the aperture mapping rules). This ptrtoint also makes the
// LDS object escape, so the compiler cannot assume it is never written.
// ---------------------------------------------------------------------------
__device__ __forceinline__ void tdm_load_tile(const __bf16* gsrc,
                                              unsigned ldsAddr) {
    typedef unsigned int u32x4 __attribute__((ext_vector_type(4)));
    typedef int          i32x8 __attribute__((ext_vector_type(8)));
    typedef int          i32x4 __attribute__((ext_vector_type(4)));

    const unsigned long long ga = (unsigned long long)(size_t)gsrc;

    u32x4 g0;
    g0.x = 1u;                                             // count=1 (valid)
    g0.y = ldsAddr;                                        // lds_addr
    g0.z = (unsigned)ga;                                   // global_addr[31:0]
    g0.w = (unsigned)((ga >> 32) & 0x01ffffffu) | (2u << 30); // [56:32] | type=2

    i32x8 g1;
    g1[0] = (1 << 16);                 // workgroup_mask=0 | data_size=1 (2B)
    g1[1] = (32 << 16);                // tensor_dim0[15:0] = 32
    g1[2] = (66 << 16);                // tensor_dim0 hi=0 | tensor_dim1 lo = 66
    g1[3] = (32 << 16);                // tensor_dim1 hi=0 | tile_dim0 = 32
    g1[4] = 66 | (3 << 16);            // tile_dim1 = 66 | tile_dim2 = 3
    g1[5] = 256;                       // tensor_dim0_stride[31:0] = 256 elems
    g1[6] = (int)(16896u << 16);       // t0s hi=0 | tensor_dim1_stride lo = 66*256
    g1[7] = 0;                         // tensor_dim1_stride hi

    i32x4 g2;
    g2[0] = 3;                         // tensor_dim2 = 3
    g2[1] = 0;                         // tensor_dim3 unused (tile_dim3=0)
    g2[2] = 16896;                     // tensor_dim2_stride = 66*256 elems
    g2[3] = 0;                         // t2s hi | tile_dim3 = 0

    i32x4 g3 = {0, 0, 0, 0};           // dim3 stride / dim4 unused

    i32x8 g4 = {0, 0, 0, 0, 0, 0, 0, 0};  // extra operand (zero-filled)

    __builtin_amdgcn_tensor_load_to_lds(g0, g1, g2, g3, g4, 0);
}

// ---------------------------------------------------------------------------
// Kernel 4: implicit-GEMM conv.
// Grid: (64 rows, 2 M-tiles, 16 batches). Block: 256 threads = 8 wave32.
// Block tile: M=128 out-ch x N=64 pixels (one image row).
// Wave tile:  32x32  -> 4 v8f accumulators, 4 WMMAs per K-step.
// K loop: 8 channel-blocks of 32 x 9 taps; TDM double-buffered staging.
// ---------------------------------------------------------------------------
__global__ __launch_bounds__(256)
void modconv_wmma_kernel(const __bf16* __restrict__ xpad,
                         const __bf16* __restrict__ Wbf,
                         float* __restrict__ out) {
    __shared__ __attribute__((aligned(64))) __bf16 tile[2][TILE_ELEMS];

    const int y  = blockIdx.x;          // image row 0..63
    const int mt = blockIdx.y;          // 0..1
    const int b  = blockIdx.z;          // 0..15

    const int tid    = threadIdx.x;
    const int lane   = tid & 31;
    const int wv     = tid >> 5;        // 0..7
    const int wvM    = wv & 3;          // 4 M positions
    const int wvN    = wv >> 2;         // 2 N positions
    const int laneLo = lane & 15;
    const bool laneHi = lane >= 16;

    const int o0 = mt * 128 + wvM * 32;     // wave's out-channel base
    const int n0 = wvN * 32;                // wave's pixel-column base

    v8f c00 = {}, c01 = {}, c10 = {}, c11 = {};

    // Tile rows p = y..y+2 (padded) == image rows y-1..y+1; cols 0..65.
    const __bf16* gRow = xpad + ((size_t)(b * PAD + y) * PAD) * IN_CH;
    const __bf16* wbatch = Wbf + (size_t)b * OUT_CH * KTAPS * IN_CH;

    const int aKoff = laneHi ? 16 : 0;   // A-frag 16-bit layout (16x32)
    const int bKoff = laneHi ? 16 : 0;   // B-frag 16-bit layout (32x16)

    // lds_addr values for the two ping-pong buffers (wave-relative offsets).
    // NOTE: these ptrtoint casts make `tile` escape into the opaque TDM ops.
    const unsigned ldsA0 = (unsigned)(size_t)(const void*)&tile[0][0];
    const unsigned ldsA1 = (unsigned)(size_t)(const void*)&tile[1][0];

    if (tid < 32) tdm_load_tile(gRow, ldsA0);   // prime buffer 0 (cb = 0)

    for (int cb = 0; cb < IN_CH / 32; ++cb) {
        if (tid < 32) {
            if (cb < IN_CH / 32 - 1) {
                // Prefetch next channel block into the ping-pong buffer.
                tdm_load_tile(gRow + (cb + 1) * 32,
                              ((cb + 1) & 1) ? ldsA1 : ldsA0);
                __builtin_amdgcn_s_wait_tensorcnt(1);  // cb's tile is resident
            } else {
                __builtin_amdgcn_s_wait_tensorcnt(0);
            }
        }
        __syncthreads();                 // tile[cb&1] visible to all waves

        // The TDM writes LDS behind the compiler's back. `tile` has escaped
        // (ptrtoint above), so a memory clobber here forces the LDS reads
        // below to be real — while keeping them in addrspace(3) => ds_load.
        asm volatile("" ::: "memory");

        const __bf16* lbuf = tile[cb & 1];
        const __bf16* wcb  = wbatch + cb * 32;

        for (int t = 0; t < KTAPS; ++t) {
            const int dy = t / 3;        // LDS row index (0..2)
            const int dx = t % 3;        // LDS col shift: image x = n + dx - 1

            // A fragments: two contiguous 16B chunks per lane per M-subtile
            const __bf16* ar0 = wcb + ((size_t)(o0      + laneLo) * KTAPS + t) * IN_CH;
            const __bf16* ar1 = wcb + ((size_t)(o0 + 16 + laneLo) * KTAPS + t) * IN_CH;
            v8bf a0lo = *(const v8bf*)(ar0 + aKoff);
            v8bf a0hi = *(const v8bf*)(ar0 + 8 + aKoff);
            v8bf a1lo = *(const v8bf*)(ar1 + aKoff);
            v8bf a1hi = *(const v8bf*)(ar1 + 8 + aKoff);
            v16bf a0 = __builtin_shufflevector(a0lo, a0hi,
                        0,1,2,3,4,5,6,7,8,9,10,11,12,13,14,15);
            v16bf a1 = __builtin_shufflevector(a1lo, a1hi,
                        0,1,2,3,4,5,6,7,8,9,10,11,12,13,14,15);

            // B fragments: 16 contiguous bf16 channels per lane from LDS
            v16bf b0 = *(const v16bf*)&lbuf[(dy * 66 + (n0      + laneLo + dx)) * 32 + bKoff];
            v16bf b1 = *(const v16bf*)&lbuf[(dy * 66 + (n0 + 16 + laneLo + dx)) * 32 + bKoff];

            c00 = __builtin_amdgcn_wmma_f32_16x16x32_bf16(false, a0, false, b0,
                                                          (short)0, c00, false, false);
            c01 = __builtin_amdgcn_wmma_f32_16x16x32_bf16(false, a0, false, b1,
                                                          (short)0, c01, false, false);
            c10 = __builtin_amdgcn_wmma_f32_16x16x32_bf16(false, a1, false, b0,
                                                          (short)0, c10, false, false);
            c11 = __builtin_amdgcn_wmma_f32_16x16x32_bf16(false, a1, false, b1,
                                                          (short)0, c11, false, false);
        }
        __syncthreads();                 // done reading tile[cb&1]
    }

    // ---- writeback: C/D f32 16x16 layout: VGPR r -> M = r + (laneHi?8:0) ----
    const int mAdd = laneHi ? 8 : 0;
    const size_t outBase = (size_t)b * OUT_CH * NPIX + (size_t)y * IMG;
    for (int r = 0; r < 8; ++r) {
        out[outBase + (size_t)(o0      + r + mAdd) * NPIX + n0      + laneLo] = c00[r];
        out[outBase + (size_t)(o0      + r + mAdd) * NPIX + n0 + 16 + laneLo] = c01[r];
        out[outBase + (size_t)(o0 + 16 + r + mAdd) * NPIX + n0      + laneLo] = c10[r];
        out[outBase + (size_t)(o0 + 16 + r + mAdd) * NPIX + n0 + 16 + laneLo] = c11[r];
    }
}

// ---------------------------------------------------------------------------
// Launch
// ---------------------------------------------------------------------------
extern "C" void kernel_launch(void* const* d_in, const int* in_sizes, int n_in,
                              void* d_out, int out_size, void* d_ws, size_t ws_size,
                              hipStream_t stream) {
    const float* x       = (const float*)d_in[0];  // [16,256,64,64]
    const float* w       = (const float*)d_in[1];  // [16,512]
    const float* style_w = (const float*)d_in[2];  // [256,512]
    const float* style_b = (const float*)d_in[3];  // [256]
    const float* weight  = (const float*)d_in[4];  // [1,256,256,3,3]
    float* out = (float*)d_out;                    // [16,256,64,64]

    // Workspace carve-up (needs ~52.6MB):
    //   [0, 16KB)                 style factors (f32)
    //   [16KB, +18.9MB)           packed bf16 weights [b][o][tap][ch]
    //   [+, +34.1MB)              zero-padded NHWC bf16 x [b][66][66][256]
    float*  sp  = (float*)d_ws;
    __bf16* Wbf = (__bf16*)((char*)d_ws + 16384);
    __bf16* xpad = (__bf16*)((char*)d_ws + 16384 +
                             (size_t)BATCH * OUT_CH * KTAPS * IN_CH * 2);

    style_factor_kernel<<<BATCH, 256, 0, stream>>>(w, style_w, style_b, sp);
    modulate_weight_kernel<<<BATCH * OUT_CH, 256, 0, stream>>>(weight, sp, Wbf);
    pack_x_kernel<<<dim3(PAD, BATCH), 256, 0, stream>>>(x, xpad);

    dim3 grid(IMG /*rows*/, OUT_CH / 128 /*M tiles*/, BATCH);
    modconv_wmma_kernel<<<grid, 256, 0, stream>>>(xpad, Wbf, out);
}
